// SwinTransformerLayer_v2_13958643712482
// MI455X (gfx1250) — compile-verified, matrix operand order checked by reading, not run on hardware
//
#include <hip/hip_runtime.h>
#include <math.h>

typedef __attribute__((ext_vector_type(16))) __bf16 v16bf;
typedef __attribute__((ext_vector_type(8)))  float  v8f;

#define NH     4
#define DIM    128
#define HD     32
#define HH     128
#define WW     128
#define NTOK   64
#define SHIFTV 4
#define LOG100 4.605170185988092f

__device__ __forceinline__ __bf16 f2bf(float x) { return (__bf16)x; }

// ---- WMMA helpers (layouts per CDNA5 ISA 7.12.2) ----

// Fragment from row-major [R x K] storage, tile rows r0..r0+15, cols k0..k0+31.
// Used for A-matrix (M x K) and for B-matrix when B = W^T with W stored [N x K].
__device__ __forceinline__ v16bf load_frag_rk(const __bf16* p, int ld, int r0, int k0, int lane) {
  int r = r0 + (lane & 15);
  const __bf16* row = p + r * ld + k0 + ((lane >> 4) << 3);
  v16bf f;
#pragma unroll
  for (int i = 0; i < 4; ++i) {
    f[2 * i]     = row[2 * i];
    f[2 * i + 1] = row[2 * i + 1];
    f[8 + 2 * i] = row[16 + 2 * i];
    f[9 + 2 * i] = row[16 + 2 * i + 1];
  }
  return f;
}

// B fragment from row-major [K x N] storage, tile rows k0..k0+31, cols n0..n0+15.
__device__ __forceinline__ v16bf load_frag_kn(const __bf16* p, int ld, int k0, int n0, int lane) {
  int n = n0 + (lane & 15);
  int kb = k0 + ((lane >> 4) << 3);
  v16bf f;
#pragma unroll
  for (int i = 0; i < 4; ++i) {
    f[2 * i]     = p[(kb + 2 * i) * ld + n];
    f[2 * i + 1] = p[(kb + 2 * i + 1) * ld + n];
    f[8 + 2 * i] = p[(kb + 16 + 2 * i) * ld + n];
    f[9 + 2 * i] = p[(kb + 17 + 2 * i) * ld + n];
  }
  return f;
}

__device__ __forceinline__ v8f wmma_bf16(v16bf a, v16bf b, v8f c) {
  return __builtin_amdgcn_wmma_f32_16x16x32_bf16(false, a, false, b, (short)0, c, false, false);
}

// ---- prep: fp32 weights -> bf16 (row-major [N x K], matches B-fragment loader) ----
__global__ void convert_weights_kernel(const float* __restrict__ qkv_w, const float* __restrict__ proj_w,
                                       const float* __restrict__ fc1_w, const float* __restrict__ fc2_w,
                                       __bf16* wqkv, __bf16* wproj, __bf16* wfc1, __bf16* wfc2) {
  const int total = 49152 + 16384 + 65536 + 65536;
  for (int i = blockIdx.x * blockDim.x + threadIdx.x; i < total; i += gridDim.x * blockDim.x) {
    if (i < 49152)       wqkv[i]           = f2bf(qkv_w[i]);
    else if (i < 65536)  wproj[i - 49152]  = f2bf(proj_w[i - 49152]);
    else if (i < 131072) wfc1[i - 65536]   = f2bf(fc1_w[i - 65536]);
    else                 wfc2[i - 131072]  = f2bf(fc2_w[i - 131072]);
  }
}

// ---- prep: continuous position bias MLP + gather + 16*sigmoid -> bias16[h][n][m] ----
__device__ __forceinline__ float rel_coord(int d) {
  float v = (float)d * (8.0f / 7.0f);
  float av = fabsf(v);
  float s = (v > 0.f) ? 1.f : ((v < 0.f) ? -1.f : 0.f);
  return s * log2f(av + 1.0f) * (1.0f / 3.0f);   // /log2(8)
}

__global__ void cpb_bias_kernel(const float* __restrict__ w1, const float* __restrict__ b1,
                                const float* __restrict__ w2, float* __restrict__ bias16) {
  __shared__ float tab[225 * NH];
  int tid = threadIdx.x;
  for (int e = tid; e < 225; e += 256) {
    int a = e / 15, b = e % 15;
    float t0 = rel_coord(a - 7), t1 = rel_coord(b - 7);
    float acc[NH] = {0.f, 0.f, 0.f, 0.f};
    for (int hd = 0; hd < 512; ++hd) {
      float hh = fmaxf(w1[hd * 2] * t0 + w1[hd * 2 + 1] * t1 + b1[hd], 0.f);
#pragma unroll
      for (int h = 0; h < NH; ++h) acc[h] += w2[h * 512 + hd] * hh;
    }
#pragma unroll
    for (int h = 0; h < NH; ++h) tab[e * NH + h] = acc[h];
  }
  __syncthreads();
  for (int i = tid; i < NH * NTOK * NTOK; i += 256) {
    int h = i >> 12, n = (i >> 6) & 63, m = i & 63;
    int dy = (n >> 3) - (m >> 3) + 7;
    int dx = (n & 7) - (m & 7) + 7;
    float v = tab[(dy * 15 + dx) * NH + h];
    bias16[i] = 16.f / (1.f + __expf(-v));
  }
}

// ---- fused Swin-v2 block: one workgroup (8 wave32) per 8x8 window ----
__global__ void __launch_bounds__(256)
swin_block_kernel(const float* __restrict__ x,
                  const float* __restrict__ n1w, const float* __restrict__ n1b,
                  const float* __restrict__ q_bias, const float* __restrict__ v_bias,
                  const float* __restrict__ logit_scale,
                  const float* __restrict__ proj_b,
                  const float* __restrict__ n2w, const float* __restrict__ n2b,
                  const float* __restrict__ fc1_b, const float* __restrict__ fc2_b,
                  const __bf16* __restrict__ wqkv, const __bf16* __restrict__ wproj,
                  const __bf16* __restrict__ wfc1, const __bf16* __restrict__ wfc2,
                  const float* __restrict__ bias16,
                  float* __restrict__ out) {
  // LDS carve (272 KB of the WGP's 320 KB)
  __shared__ float lds[69632];
  __shared__ int   sCode[NTOK];
  __shared__ float sMu[NTOK], sRs[NTOK];
  float*  XF  = lds;                               //  64x128 f32   (x window / x1 residual)
  __bf16* X16 = (__bf16*)(lds + 8192);             //  64x128 bf16  (A for qkv, later x1)
  float*  SCR = lds + 12288;                       //  up to 64x512 f32 scratch
  __bf16* Q16 = (__bf16*)(lds + 45056);            //  64x128 bf16  q-hat (scale folded)
  __bf16* K16 = (__bf16*)(lds + 49152);            //  64x128 bf16  k-hat
  __bf16* V16 = (__bf16*)(lds + 53248);            //  64x128 bf16  v
  __bf16* P16 = (__bf16*)(lds + 57344);            //  4x64x64 bf16 attn probs
  __bf16* A16 = (__bf16*)(lds + 65536);            //  64x128 bf16  attn out
  __bf16* H16 = (__bf16*)(lds + 45056);            //  64x512 bf16  MLP hidden (reuses Q/K/V/P)

  const int tid = threadIdx.x;
  const int wave = tid >> 5, lane = tid & 31;
  const int b  = blockIdx.x >> 8;
  const int wy = (blockIdx.x >> 4) & 15;
  const int wx = blockIdx.x & 15;

  // 1) gather shifted window of x (B,C,H,W layout) straight into LDS with
  //    CDNA5 async global->LDS copies (ASYNCcnt-tracked, no VGPR staging)
  for (int i = tid; i < NTOK * DIM; i += 256) {
    int c = i >> 6, t = i & 63;
    int hsrc = ((wy << 3) + (t >> 3) + SHIFTV) & 127;
    int wsrc = ((wx << 3) + (t & 7) + SHIFTV) & 127;
    const float* gsrc = x + (((size_t)b * DIM + c) * HH + hsrc) * WW + wsrc;
    // low 32 bits of a generic shared pointer are the LDS byte offset
    unsigned ldsoff = (unsigned)(unsigned long long)(void*)(XF + t * DIM + c);
    asm volatile("global_load_async_to_lds_b32 %0, %1, off"
                 :: "v"(ldsoff), "v"(gsrc) : "memory");
  }
  asm volatile("s_wait_asynccnt 0x0" ::: "memory");
  if (tid < NTOK) {  // shift-mask region code per token
    int hs = (wy << 3) + (tid >> 3), ws2 = (wx << 3) + (tid & 7);
    int rh = hs < 120 ? 0 : (hs < 124 ? 1 : 2);
    int rw = ws2 < 120 ? 0 : (ws2 < 124 ? 1 : 2);
    sCode[tid] = rh * 3 + rw;
  }
  __syncthreads();
  for (int i = tid; i < NTOK * DIM; i += 256) X16[i] = f2bf(XF[i]);
  __syncthreads();

  // 2) qkv = x @ qkv_w^T + bias   (64x128)x(128x384): 4 M-tiles x 24 N-tiles x 4 K
  {
    int mt = wave & 3;
    int ng0 = (wave >> 2) * 12;
    v16bf afr[4];
#pragma unroll
    for (int kk = 0; kk < 4; ++kk) afr[kk] = load_frag_rk(X16, DIM, mt * 16, kk * 32, lane);
    for (int j = 0; j < 12; ++j) {
      int n0 = (ng0 + j) * 16;
      int col = n0 + (lane & 15);
      float bv = (col < 128) ? q_bias[col] : (col < 256 ? 0.f : v_bias[col - 256]);
      v16bf bfr[4];
#pragma unroll
      for (int kk = 0; kk < 4; ++kk) bfr[kk] = load_frag_rk(wqkv, DIM, n0, kk * 32, lane);
      v8f acc = {bv, bv, bv, bv, bv, bv, bv, bv};
#pragma unroll
      for (int kk = 0; kk < 4; ++kk) acc = wmma_bf16(afr[kk], bfr[kk], acc);
      int mb = mt * 16 + ((lane >> 4) << 3);
#pragma unroll
      for (int r = 0; r < 8; ++r) SCR[(mb + r) * 384 + col] = acc[r];
    }
  }
  __syncthreads();

  // 3) cosine-normalize q,k (fold exp(min(logit_scale,log100)) into q), copy v -> bf16
  {
    int t = tid >> 2, h = tid & 3;
    const float* qrow = SCR + t * 384 + h * HD;
    const float* krow = qrow + 128;
    float sq = 0.f, sk = 0.f;
#pragma unroll
    for (int d = 0; d < HD; ++d) { sq += qrow[d] * qrow[d]; sk += krow[d] * krow[d]; }
    float iq = 1.f / fmaxf(sqrtf(sq), 1e-12f);
    float ik = 1.f / fmaxf(sqrtf(sk), 1e-12f);
    iq *= __expf(fminf(logit_scale[h], LOG100));
#pragma unroll
    for (int d = 0; d < HD; ++d) {
      Q16[t * DIM + h * HD + d] = f2bf(qrow[d] * iq);
      K16[t * DIM + h * HD + d] = f2bf(krow[d] * ik);
    }
  }
  for (int i = tid; i < NTOK * DIM; i += 256)
    V16[i] = f2bf(SCR[(i >> 7) * 384 + 256 + (i & 127)]);
  __syncthreads();

  // 4) scores = q-hat @ k-hat^T + 16*sigmoid(cpb) + shift mask    (per head 64x64, K=32)
  for (int jb = wave; jb < 64; jb += 8) {
    int h = jb >> 4, i = (jb >> 2) & 3, j = jb & 3;
    v16bf a = load_frag_rk(Q16, DIM, i * 16, h * HD, lane);
    v16bf bfr = load_frag_rk(K16, DIM, j * 16, h * HD, lane);
    v8f acc = {};
    acc = wmma_bf16(a, bfr, acc);
    int m = j * 16 + (lane & 15);
    int nb = i * 16 + ((lane >> 4) << 3);
    int cm = sCode[m];
#pragma unroll
    for (int r = 0; r < 8; ++r) {
      int n = nb + r;
      float v = acc[r] + bias16[(h << 12) + (n << 6) + m];
      if (sCode[n] != cm) v -= 100.f;
      SCR[(h << 12) + (n << 6) + m] = v;
    }
  }
  __syncthreads();

  // 5) softmax rows -> bf16 probs
  {
    int h = tid >> 6, n = tid & 63;
    float* row = SCR + (h << 12) + (n << 6);
    float mx = row[0];
#pragma unroll
    for (int m = 1; m < 64; ++m) mx = fmaxf(mx, row[m]);
    float s = 0.f;
#pragma unroll
    for (int m = 0; m < 64; ++m) { float e = __expf(row[m] - mx); row[m] = e; s += e; }
    float inv = 1.f / s;
#pragma unroll
    for (int m = 0; m < 64; ++m) P16[(h << 12) + (n << 6) + m] = f2bf(row[m] * inv);
  }
  __syncthreads();

  // 6) attn_out = P @ V  (per head 64x64 @ 64x32)
  for (int jb = wave; jb < 32; jb += 8) {
    int h = jb >> 3, i = (jb >> 1) & 3, j = jb & 1;
    v8f acc = {};
#pragma unroll
    for (int kk = 0; kk < 2; ++kk) {
      v16bf a = load_frag_rk(P16 + (h << 12), 64, i * 16, kk * 32, lane);
      v16bf bfr = load_frag_kn(V16, DIM, kk * 32, h * HD + j * 16, lane);
      acc = wmma_bf16(a, bfr, acc);
    }
    int col = h * HD + j * 16 + (lane & 15);
    int mb = i * 16 + ((lane >> 4) << 3);
#pragma unroll
    for (int r = 0; r < 8; ++r) A16[(mb + r) * DIM + col] = f2bf(acc[r]);
  }
  __syncthreads();

  // 7) y = attn_out @ proj_w^T + proj_b
  for (int jb = wave; jb < 32; jb += 8) {
    int i = jb >> 3, n0 = (jb & 7) * 16;
    int col = n0 + (lane & 15);
    float bv = proj_b[col];
    v16bf bfr[4];
#pragma unroll
    for (int kk = 0; kk < 4; ++kk) bfr[kk] = load_frag_rk(wproj, DIM, n0, kk * 32, lane);
    v8f acc = {bv, bv, bv, bv, bv, bv, bv, bv};
#pragma unroll
    for (int kk = 0; kk < 4; ++kk)
      acc = wmma_bf16(load_frag_rk(A16, DIM, i * 16, kk * 32, lane), bfr[kk], acc);
    int mb = i * 16 + ((lane >> 4) << 3);
#pragma unroll
    for (int r = 0; r < 8; ++r) SCR[(mb + r) * DIM + col] = acc[r];
  }
  __syncthreads();

  // 8) x1 = shortcut + LN(y)  (post-norm residual): stats then coalesced apply
  if (tid < NTOK) {
    float mu = 0.f;
    for (int c = 0; c < DIM; ++c) mu += SCR[tid * DIM + c];
    mu *= (1.f / DIM);
    float var = 0.f;
    for (int c = 0; c < DIM; ++c) { float d = SCR[tid * DIM + c] - mu; var += d * d; }
    sMu[tid] = mu;
    sRs[tid] = rsqrtf(var * (1.f / DIM) + 1e-5f);
  }
  __syncthreads();
  for (int i = tid; i < NTOK * DIM; i += 256) {
    int t = i >> 7, c = i & 127;
    float x1 = XF[i] + (SCR[i] - sMu[t]) * sRs[t] * n1w[c] + n1b[c];
    XF[i] = x1;
    X16[i] = f2bf(x1);
  }
  __syncthreads();

  // 9) h = silu(x1 @ fc1_w^T + fc1_b)   (64x128)x(128x512)
  for (int jb = wave; jb < 128; jb += 8) {
    int i = jb >> 5, n0 = (jb & 31) * 16;
    int col = n0 + (lane & 15);
    float bv = fc1_b[col];
    v16bf bfr[4];
#pragma unroll
    for (int kk = 0; kk < 4; ++kk) bfr[kk] = load_frag_rk(wfc1, DIM, n0, kk * 32, lane);
    v8f acc = {bv, bv, bv, bv, bv, bv, bv, bv};
#pragma unroll
    for (int kk = 0; kk < 4; ++kk)
      acc = wmma_bf16(load_frag_rk(X16, DIM, i * 16, kk * 32, lane), bfr[kk], acc);
    int mb = i * 16 + ((lane >> 4) << 3);
#pragma unroll
    for (int r = 0; r < 8; ++r) {
      float v = acc[r];
      v = v / (1.f + __expf(-v));
      H16[(mb + r) * 512 + col] = f2bf(v);
    }
  }
  __syncthreads();

  // 10) z = h @ fc2_w^T + fc2_b   (64x512)x(512x128), k-chunks of 4 with B prefetch
  for (int jb = wave; jb < 32; jb += 8) {
    int i = jb >> 3, n0 = (jb & 7) * 16;
    int col = n0 + (lane & 15);
    float bv = fc2_b[col];
    v8f acc = {bv, bv, bv, bv, bv, bv, bv, bv};
#pragma unroll
    for (int kc = 0; kc < 4; ++kc) {
      v16bf bfr[4];
#pragma unroll
      for (int kk = 0; kk < 4; ++kk)
        bfr[kk] = load_frag_rk(wfc2, 512, n0, (kc * 4 + kk) * 32, lane);
#pragma unroll
      for (int kk = 0; kk < 4; ++kk)
        acc = wmma_bf16(load_frag_rk(H16, 512, i * 16, (kc * 4 + kk) * 32, lane), bfr[kk], acc);
    }
    int mb = i * 16 + ((lane >> 4) << 3);
#pragma unroll
    for (int r = 0; r < 8; ++r) SCR[(mb + r) * DIM + col] = acc[r];
  }
  __syncthreads();

  // 11) x2 = x1 + LN(z); stats then coalesced scatter back to (B,C,H,W)
  if (tid < NTOK) {
    float mu = 0.f;
    for (int c = 0; c < DIM; ++c) mu += SCR[tid * DIM + c];
    mu *= (1.f / DIM);
    float var = 0.f;
    for (int c = 0; c < DIM; ++c) { float d = SCR[tid * DIM + c] - mu; var += d * d; }
    sMu[tid] = mu;
    sRs[tid] = rsqrtf(var * (1.f / DIM) + 1e-5f);
  }
  __syncthreads();
  for (int i = tid; i < NTOK * DIM; i += 256) {
    int c = i >> 6, t = i & 63;
    int hdst = ((wy << 3) + (t >> 3) + SHIFTV) & 127;
    int wdst = ((wx << 3) + (t & 7) + SHIFTV) & 127;
    float v = XF[t * DIM + c] + (SCR[t * DIM + c] - sMu[t]) * sRs[t] * n2w[c] + n2b[c];
    out[(((size_t)b * DIM + c) * HH + hdst) * WW + wdst] = v;
  }
}

extern "C" void kernel_launch(void* const* d_in, const int* in_sizes, int n_in,
                              void* d_out, int out_size, void* d_ws, size_t ws_size,
                              hipStream_t stream) {
  const float* x           = (const float*)d_in[0];
  const float* n1w         = (const float*)d_in[1];
  const float* n1b         = (const float*)d_in[2];
  const float* qkv_w       = (const float*)d_in[3];
  const float* q_bias      = (const float*)d_in[4];
  const float* v_bias      = (const float*)d_in[5];
  const float* logit_scale = (const float*)d_in[6];
  const float* cpb_w1      = (const float*)d_in[7];
  const float* cpb_b1      = (const float*)d_in[8];
  const float* cpb_w2      = (const float*)d_in[9];
  const float* proj_w      = (const float*)d_in[10];
  const float* proj_b      = (const float*)d_in[11];
  const float* n2w         = (const float*)d_in[12];
  const float* n2b         = (const float*)d_in[13];
  const float* fc1_w       = (const float*)d_in[14];
  const float* fc1_b       = (const float*)d_in[15];
  const float* fc2_w       = (const float*)d_in[16];
  const float* fc2_b       = (const float*)d_in[17];

  char* ws = (char*)d_ws;
  __bf16* wqkv  = (__bf16*)(ws);             // 384x128 bf16
  __bf16* wproj = (__bf16*)(ws + 98304);     // 128x128 bf16
  __bf16* wfc1  = (__bf16*)(ws + 131072);    // 512x128 bf16
  __bf16* wfc2  = (__bf16*)(ws + 262144);    // 128x512 bf16
  float*  bias16 = (float*)(ws + 393216);    // 4x64x64 f32

  convert_weights_kernel<<<192, 256, 0, stream>>>(qkv_w, proj_w, fc1_w, fc2_w,
                                                  wqkv, wproj, wfc1, wfc2);
  cpb_bias_kernel<<<1, 256, 0, stream>>>(cpb_w1, cpb_b1, cpb_w2, bias16);
  swin_block_kernel<<<4096, 256, 0, stream>>>(x, n1w, n1b, q_bias, v_bias, logit_scale,
                                              proj_b, n2w, n2b, fc1_b, fc2_b,
                                              wqkv, wproj, wfc1, wfc2, bias16,
                                              (float*)d_out);
}
